// TimeMixV4_54047868453534
// MI455X (gfx1250) — matrix operand order repaired and checked by reading, not compiled, not run on hardware
//
#include <hip/hip_runtime.h>

#define H_DIM 2048
#define B_DIM 4
#define T_DIM 2048

typedef __bf16 bf16;
typedef __attribute__((ext_vector_type(16))) __bf16 v16bf;
typedef __attribute__((ext_vector_type(8)))  __bf16 v8bf;
typedef __attribute__((ext_vector_type(8)))  float   v8f;

__device__ __forceinline__ bf16 f2bf(float x) { return (bf16)x; }

// --- gfx1250 async global->LDS copies (ASYNCcnt-tracked); inst offset applies
// to BOTH the LDS and the global address (ISA 15.18.3). -----------------------
__device__ __forceinline__ void async_b128(unsigned lds, unsigned long long ga) {
    asm volatile("global_load_async_to_lds_b128 %0, %1, off"
                 :: "v"(lds), "v"(ga) : "memory");
}
__device__ __forceinline__ void async_b128_o16(unsigned lds, unsigned long long ga) {
    asm volatile("global_load_async_to_lds_b128 %0, %1, off offset:16"
                 :: "v"(lds), "v"(ga) : "memory");
}
__device__ __forceinline__ void async_b128_o32(unsigned lds, unsigned long long ga) {
    asm volatile("global_load_async_to_lds_b128 %0, %1, off offset:32"
                 :: "v"(lds), "v"(ga) : "memory");
}
__device__ __forceinline__ void async_b128_o48(unsigned lds, unsigned long long ga) {
    asm volatile("global_load_async_to_lds_b128 %0, %1, off offset:48"
                 :: "v"(lds), "v"(ga) : "memory");
}
__device__ __forceinline__ void wait_async0() {
    asm volatile("s_wait_asynccnt 0x0" ::: "memory");
}

// ---------------------------------------------------------------------------
// Kernel 1: time-shift mixing  (x, x_prev) -> xk, xv, xr  (bf16)
// ---------------------------------------------------------------------------
__global__ __launch_bounds__(256) void mix_kernel(
    const float* __restrict__ x,
    const float* __restrict__ tmk, const float* __restrict__ tmv,
    const float* __restrict__ tmr,
    bf16* __restrict__ xk, bf16* __restrict__ xv, bf16* __restrict__ xr)
{
    size_t base = ((size_t)blockIdx.x * 256 + threadIdx.x) * 8;
    int h = (int)(base & (H_DIM - 1));
    int t = (int)((base >> 11) & (T_DIM - 1));

    float c[8], p[8], mk[8], mv[8], mr[8];
    *(float4*)&c[0] = *(const float4*)(x + base);
    *(float4*)&c[4] = *(const float4*)(x + base + 4);
    if (t != 0) {
        *(float4*)&p[0] = *(const float4*)(x + base - H_DIM);
        *(float4*)&p[4] = *(const float4*)(x + base - H_DIM + 4);
    } else {
#pragma unroll
        for (int i = 0; i < 8; ++i) p[i] = 0.0f;
    }
    *(float4*)&mk[0] = *(const float4*)(tmk + h);
    *(float4*)&mk[4] = *(const float4*)(tmk + h + 4);
    *(float4*)&mv[0] = *(const float4*)(tmv + h);
    *(float4*)&mv[4] = *(const float4*)(tmv + h + 4);
    *(float4*)&mr[0] = *(const float4*)(tmr + h);
    *(float4*)&mr[4] = *(const float4*)(tmr + h + 4);

    v8bf ok, ov, orr;
#pragma unroll
    for (int i = 0; i < 8; ++i) {
        float d = c[i] - p[i];
        ok[i]  = f2bf(p[i] + d * mk[i]);
        ov[i]  = f2bf(p[i] + d * mv[i]);
        orr[i] = f2bf(p[i] + d * mr[i]);
    }
    *(v8bf*)(xk + base) = ok;
    *(v8bf*)(xv + base) = ov;
    *(v8bf*)(xr + base) = orr;
}

// ---------------------------------------------------------------------------
// Kernel 2: fused transpose + fp32->bf16:  W[N][K] -> Wt[K][N]
// ---------------------------------------------------------------------------
__global__ __launch_bounds__(256) void transpose_cvt_kernel(
    const float* __restrict__ W, bf16* __restrict__ Wt, int N, int K)
{
    __shared__ float tile[64][65];
    const int bk = blockIdx.x * 64;
    const int bn = blockIdx.y * 64;
    const int t  = threadIdx.x;
    const int r  = t >> 2;            // 0..63
    const int c  = (t & 3) * 16;      // 0,16,32,48

    const float* src = W + (size_t)(bn + r) * K + bk + c;
#pragma unroll
    for (int i = 0; i < 16; i += 4)
        *(float4*)&tile[r][c + i] = *(const float4*)(src + i);

    __syncthreads();

    bf16* dst = Wt + (size_t)(bk + r) * N + bn + c;
#pragma unroll
    for (int i = 0; i < 16; i += 8) {
        v8bf o;
#pragma unroll
        for (int j = 0; j < 8; ++j) o[j] = f2bf(tile[c + i + j][r]);
        *(v8bf*)(dst + i) = o;
    }
}

// ---------------------------------------------------------------------------
// Kernel 3: C[M,N] = A[M,K](bf16) * Wt[K,N](bf16) + bias[N]
// 256 threads (8 waves), tile 256(M) x 128(N), BK=32.
// Each wave: two 16-row strips x 8 N-tiles = 16 WMMA / K-step, B frags reused.
// Async global->LDS staging, double-buffered LDS, K-loop unrolled x2 so the
// buffer indices are compile-time constants.
// ---------------------------------------------------------------------------
#define AP 40    // padded A row (halfs): 20 dwords -> conflict-free frag reads
#define BP 136   // padded B row (halfs): 68 dwords -> worst-case 2-way

#define GEMM_STAGE(BUF)                          \
    {                                            \
        async_b128(ldsA[BUF], ga);               \
        async_b128_o16(ldsA[BUF], ga);           \
        async_b128_o32(ldsA[BUF], ga);           \
        async_b128_o48(ldsA[BUF], ga);           \
        async_b128(ldsB[BUF], gb);               \
        async_b128_o16(ldsB[BUF], gb);           \
        ga += astep;                             \
        gb += bstep;                             \
    }

#define GEMM_COMPUTE(BUF)                                                      \
    {                                                                          \
        const bf16 (*Ac)[AP] = As[BUF];                                        \
        const bf16 (*Bc)[BP] = Bs[BUF];                                        \
        v8bf alo0 = *(const v8bf*)&Ac[mrow0][kb];                              \
        v8bf ahi0 = *(const v8bf*)&Ac[mrow0][kb + 16];                         \
        v8bf alo1 = *(const v8bf*)&Ac[mrow1][kb];                              \
        v8bf ahi1 = *(const v8bf*)&Ac[mrow1][kb + 16];                         \
        v16bf af0 = __builtin_shufflevector(alo0, ahi0,                        \
            0, 1, 2, 3, 4, 5, 6, 7, 8, 9, 10, 11, 12, 13, 14, 15);             \
        v16bf af1 = __builtin_shufflevector(alo1, ahi1,                        \
            0, 1, 2, 3, 4, 5, 6, 7, 8, 9, 10, 11, 12, 13, 14, 15);             \
        _Pragma("unroll")                                                      \
        for (int nt = 0; nt < 8; ++nt) {                                       \
            v16bf bfr;                                                         \
            _Pragma("unroll")                                                  \
            for (int s = 0; s < 16; ++s)                                       \
                bfr[s] = Bc[kb + (s & 7) + ((s >> 3) << 4)][nt * 16 + (lane & 15)]; \
            acc0[nt] = __builtin_amdgcn_wmma_f32_16x16x32_bf16(                \
                false, af0, false, bfr, (short)0, acc0[nt], false, false);     \
            acc1[nt] = __builtin_amdgcn_wmma_f32_16x16x32_bf16(                \
                false, af1, false, bfr, (short)0, acc1[nt], false, false);     \
        }                                                                      \
    }

__global__ __launch_bounds__(256) void gemm_kernel(
    const bf16* __restrict__ A,    // [M,K]
    const bf16* __restrict__ Wt,   // [K,N]
    const float* __restrict__ bias,
    float* __restrict__ C,         // [M,N]
    int M, int N, int K)           // K % 64 == 0
{
    __shared__ __align__(64) bf16 As[2][256][AP];
    __shared__ __align__(64) bf16 Bs[2][32][BP];

    const int tid  = threadIdx.x;
    const int wave = tid >> 5;
    const int lane = tid & 31;
    const int bm = blockIdx.y * 256;
    const int bn = blockIdx.x * 128;

    v8f acc0[8], acc1[8];
#pragma unroll
    for (int i = 0; i < 8; ++i)
#pragma unroll
        for (int j = 0; j < 8; ++j) { acc0[i][j] = 0.0f; acc1[i][j] = 0.0f; }

    // staging: one thread = one 64B A row slice; 16 halfs of one Bs row
    const int am = tid;                 // 0..255
    const int kr = tid >> 3;            // 0..31   Bs row (K)
    const int nc = (tid & 7) * 16;      // 0..112  Bs column chunk

    unsigned long long ga = (unsigned long long)(A  + (size_t)(bm + am) * K);
    unsigned long long gb = (unsigned long long)(Wt + (size_t)kr * N + bn + nc);
    const unsigned long long astep = 32ull * sizeof(bf16);
    const unsigned long long bstep = 32ull * (unsigned long long)N * sizeof(bf16);

    const unsigned ldsA[2] = { (unsigned)(unsigned long long)&As[0][am][0],
                               (unsigned)(unsigned long long)&As[1][am][0] };
    const unsigned ldsB[2] = { (unsigned)(unsigned long long)&Bs[0][kr][nc],
                               (unsigned)(unsigned long long)&Bs[1][kr][nc] };

    // A-fragment addressing (16-bit A 16x32 layout), two strips per wave
    const int mrow0 = wave * 32 + (lane & 15);
    const int mrow1 = mrow0 + 16;
    const int kb    = (lane >> 4) * 8;

    // prologue: slice 0 -> buffer 0
    GEMM_STAGE(0);

    for (int k0 = 0; k0 < K; k0 += 64) {
        // ---- phase 0: compute buf0, prefetch next slice into buf1 ----
        wait_async0();
        __syncthreads();
        GEMM_STAGE(1);                    // k0+32 < K always (K % 64 == 0)
        GEMM_COMPUTE(0);

        // ---- phase 1: compute buf1, prefetch into buf0 ----
        wait_async0();
        __syncthreads();
        if (k0 + 64 < K) GEMM_STAGE(0);   // uniform
        GEMM_COMPUTE(1);
    }

    // epilogue: lane 0-15: M=r, N=lane ; lane 16-31: M=8+r, N=lane-16
    const int mb0 = bm + wave * 32 + ((lane >> 4) * 8);
    const int mb1 = mb0 + 16;
#pragma unroll
    for (int nt = 0; nt < 8; ++nt) {
        int n = bn + nt * 16 + (lane & 15);
        float bv = bias[n];
#pragma unroll
        for (int r = 0; r < 8; ++r) {
            C[(size_t)(mb0 + r) * N + n] = acc0[nt][r] + bv;
            C[(size_t)(mb1 + r) * N + n] = acc1[nt][r] + bv;
        }
    }
}

// ---------------------------------------------------------------------------
// Kernel 4: sequential WKV scan over T + sigmoid(r) gating -> rwkv (bf16)
// ---------------------------------------------------------------------------
__global__ __launch_bounds__(256) void wkv_kernel(
    const float* __restrict__ kf, const float* __restrict__ vf,
    const float* __restrict__ rf,
    const float* __restrict__ time_decay, const float* __restrict__ time_first,
    bf16* __restrict__ rwkv)
{
    int idx = blockIdx.x * 256 + threadIdx.x;   // < B*H
    int h = idx & (H_DIM - 1);
    int b = idx >> 11;
    const float decay = -__expf(time_decay[h]);
    const float bonus = time_first[h];

    float aa = 0.0f, bb = 0.0f, pp = -1e30f;
    size_t base = (size_t)b * T_DIM * H_DIM + h;
    for (int t = 0; t < T_DIM; ++t) {
        size_t o = base + (size_t)t * H_DIM;
        float kt = kf[o];
        float vt = vf[o];

        float ww = bonus + kt;
        float p  = fmaxf(pp, ww);
        float e1 = __expf(pp - p);
        float e2 = __expf(ww - p);
        float out = (e1 * aa + e2 * vt) / (e1 * bb + e2);

        float ww2 = pp + decay;
        float p2  = fmaxf(ww2, kt);
        e1 = __expf(ww2 - p2);
        e2 = __expf(kt - p2);
        aa = e1 * aa + e2 * vt;
        bb = e1 * bb + e2;
        pp = p2;

        float rt  = rf[o];
        float sig = 1.0f / (1.0f + __expf(-rt));
        rwkv[o] = f2bf(sig * out);
    }
}

// ---------------------------------------------------------------------------
extern "C" void kernel_launch(void* const* d_in, const int* in_sizes, int n_in,
                              void* d_out, int out_size, void* d_ws, size_t ws_size,
                              hipStream_t stream)
{
    const float* x   = (const float*)d_in[0];
    const float* Wk  = (const float*)d_in[1];
    const float* bk  = (const float*)d_in[2];
    const float* Wv  = (const float*)d_in[3];
    const float* bv  = (const float*)d_in[4];
    const float* Wr  = (const float*)d_in[5];
    const float* br  = (const float*)d_in[6];
    const float* Wo  = (const float*)d_in[7];
    const float* bo  = (const float*)d_in[8];
    const float* tmk = (const float*)d_in[9];
    const float* tmv = (const float*)d_in[10];
    const float* tmr = (const float*)d_in[11];
    const float* td  = (const float*)d_in[12];
    const float* tf  = (const float*)d_in[13];

    const size_t MK = (size_t)B_DIM * T_DIM * H_DIM;   // 16,777,216
    const size_t HH = (size_t)H_DIM * H_DIM;           //  4,194,304
    const int M = B_DIM * T_DIM;                       //  8192

    // workspace layout
    bf16*  xk   = (bf16*)d_ws;
    bf16*  xv   = xk + MK;
    bf16*  xr   = xv + MK;
    bf16*  wkt  = xr + MK;       // transposed bf16 weights [K][N]
    bf16*  wvt  = wkt + HH;
    bf16*  wrt  = wvt + HH;
    bf16*  wot  = wrt + HH;
    float* kf   = (float*)(wot + HH);
    float* vf   = kf + MK;
    float* rf   = vf + MK;
    bf16*  rwkv = (bf16*)(rf + MK);

    // 1) mixing + bf16 conversion (one block per (b,t) row)
    mix_kernel<<<(unsigned)(MK / 2048), 256, 0, stream>>>(x, tmk, tmv, tmr, xk, xv, xr);

    // 2) weight transpose + conversion
    dim3 gt(H_DIM / 64, H_DIM / 64);   // (32, 32)
    transpose_cvt_kernel<<<gt, 256, 0, stream>>>(Wk, wkt, H_DIM, H_DIM);
    transpose_cvt_kernel<<<gt, 256, 0, stream>>>(Wv, wvt, H_DIM, H_DIM);
    transpose_cvt_kernel<<<gt, 256, 0, stream>>>(Wr, wrt, H_DIM, H_DIM);
    transpose_cvt_kernel<<<gt, 256, 0, stream>>>(Wo, wot, H_DIM, H_DIM);

    // 3) k / v / r projections (WMMA bf16 GEMMs, 256x128 tiles)
    dim3 gg(H_DIM / 128, M / 256);     // (16, 32)
    gemm_kernel<<<gg, 256, 0, stream>>>(xk, wkt, bk, kf, M, H_DIM, H_DIM);
    gemm_kernel<<<gg, 256, 0, stream>>>(xv, wvt, bv, vf, M, H_DIM, H_DIM);
    gemm_kernel<<<gg, 256, 0, stream>>>(xr, wrt, br, rf, M, H_DIM, H_DIM);

    // 4) WKV scan + sigmoid gating
    wkv_kernel<<<(B_DIM * H_DIM) / 256, 256, 0, stream>>>(kf, vf, rf, td, tf, rwkv);

    // 5) output projection
    gemm_kernel<<<gg, 256, 0, stream>>>(rwkv, wot, bo, (float*)d_out, M, H_DIM, H_DIM);
}
